// GRIL_26070451487093
// MI455X (gfx1250) — matrix operand order repaired and checked by reading, not compiled, not run on hardware
//
#include <hip/hip_runtime.h>
#include <hip/hip_bf16.h>
#include <math.h>

// ---------------------------------------------------------------------------
// GRIL cell on gfx1250: f16 WMMA GEMMs (f32 accum) for graph aggregation and
// gate convolutions; everything resident in L2 (~30 MB workspace).
// GEMM inner loop: all five fragments (a,b0..b3) load as one clause, then a
// sched_barrier(0) pins the loads ahead of the WMMA group so the compiler
// cannot sink/coalesce them -> staggered s_wait_loadcnt (6/4/2/0) instead of
// four full stalls, with a one-generation live set (no scratch spills).
// ---------------------------------------------------------------------------

typedef __attribute__((ext_vector_type(16))) _Float16 v16h;
typedef __attribute__((ext_vector_type(8)))  _Float16 v8h;
typedef __attribute__((ext_vector_type(8)))  float    v8f;

#define NN        1024      // N_NODES
#define HID       64        // HIDDEN
#define CIN       2         // INPUT_SIZE
#define BT        8         // BATCH
#define STEPS     32
#define CGATE     66        // CIN + HID
#define NSUPP     4         // 2*ORDER
#define CCONV     330       // (NSUPP+1)*CGATE
#define KPAD      352       // CCONV padded to multiple of 32
#define BCROWS    528       // BT*CGATE
#define IMP_SIZE  (BT*CIN*NN*STEPS)   // 524288

// ---------------- workspace layout (bytes, all 256-aligned) ----------------
#define OFF_SUPPH   ((size_t)0)                       // 4*1024*1024 f16 = 8 MB
#define OFF_HSTATE  ((size_t)8388608)                 // 8*64*1024 f32   = 2 MB
#define OFF_ZGA     ((size_t)10485760)                // 528*1024 f16
#define OFF_ZCA     ((size_t)11567104)                // 528*1024 f16
#define OFF_ZCATG   ((size_t)12648448)                // 8*1024*352 f16 = 5.5 MB
#define OFF_ZCATC   ((size_t)18415616)                // 8*1024*352 f16
#define OFF_R       ((size_t)24182784)                // 8*64*1024 f32
#define OFF_U       ((size_t)26279936)                // 8*64*1024 f32
#define OFF_CC      ((size_t)28377088)                // 8*64*1024 f32
#define OFF_WPAD    ((size_t)30474240)                // 3*64*352 f16
#define OFF_ROWS    ((size_t)30609408)                // 1024 f32
#define OFF_COLS    ((size_t)30613504)                // 1024 f32
// total ~30.6 MB

#if defined(__has_builtin)
#if __has_builtin(__builtin_amdgcn_sched_barrier)
#define SCHED_FENCE() __builtin_amdgcn_sched_barrier(0)
#endif
#endif
#ifndef SCHED_FENCE
#define SCHED_FENCE()
#endif

// ------------------------- WMMA fragment loaders ---------------------------
// A (16x32 f16, row-major source): lanes 0-15 rows m0..m0+15 with K kb=0,
// lanes 16-31 same rows with kb=8; halfs 0..7 -> K=kb+0..7, 8..15 -> kb+16..23.
__device__ __forceinline__ v16h load_fragA_row(const _Float16* A, int lda,
                                               int m0, int k0) {
  int lane = threadIdx.x & 31;
  int m  = m0 + (lane & 15);
  int kb = (lane & 16) ? 8 : 0;
  const _Float16* p = A + (size_t)m * lda + k0 + kb;
  v8h lo = *(const v8h*)p;
  v8h hi = *(const v8h*)(p + 16);
  v16h f;
#pragma unroll
  for (int j = 0; j < 8; ++j) { f[j] = lo[j]; f[8 + j] = hi[j]; }
  return f;
}

// B (32x16 f16) from col-major storage Bc[n][k]: lane n = n0+(lane&15),
// lanes 0-15 kb=0, lanes 16-31 kb=8; halfs 0..7 -> K=kb+0..7, 8..15 -> kb+16..23.
__device__ __forceinline__ v16h load_fragB_col(const _Float16* Bc, int ldb,
                                               int n0, int k0) {
  int lane = threadIdx.x & 31;
  int n  = n0 + (lane & 15);
  int kb = (lane & 16) ? 8 : 0;
  const _Float16* p = Bc + (size_t)n * ldb + k0 + kb;
  v8h lo = *(const v8h*)p;
  v8h hi = *(const v8h*)(p + 16);
  v16h f;
#pragma unroll
  for (int j = 0; j < 8; ++j) { f[j] = lo[j]; f[8 + j] = hi[j]; }
  return f;
}

__device__ __forceinline__ v8f wmma16(v16h a, v16h b, v8f c) {
  return __builtin_amdgcn_wmma_f32_16x16x32_f16(false, a, false, b,
                                                (short)0, c, false, false);
}

// 16xK x Kx64 strip GEMM: acc[0..3] are the four 16x16 output tiles at
// columns n0..n0+48. Loads pinned ahead of WMMAs by sched_barrier so waits
// stagger; unroll 1 keeps the live set to one fragment generation.
__device__ __forceinline__ void gemm_strip(const _Float16* __restrict__ A, int lda,
                                           const _Float16* __restrict__ Bc, int ldb,
                                           int m0, int n0, int K, v8f acc[4]) {
#pragma unroll
  for (int j = 0; j < 4; ++j)
#pragma unroll
    for (int e = 0; e < 8; ++e) acc[j][e] = 0.0f;

#pragma unroll 1
  for (int k0 = 0; k0 < K; k0 += 32) {
    v16h a  = load_fragA_row(A, lda, m0, k0);
    v16h b0 = load_fragB_col(Bc, ldb, n0,      k0);
    v16h b1 = load_fragB_col(Bc, ldb, n0 + 16, k0);
    v16h b2 = load_fragB_col(Bc, ldb, n0 + 32, k0);
    v16h b3 = load_fragB_col(Bc, ldb, n0 + 48, k0);
    SCHED_FENCE();                  // keep all 10 loads ahead of the WMMAs
    acc[0] = wmma16(a, b0, acc[0]);
    acc[1] = wmma16(a, b1, acc[1]);
    acc[2] = wmma16(a, b2, acc[2]);
    acc[3] = wmma16(a, b3, acc[3]);
  }
}

// ------------------------------ init kernels -------------------------------
__global__ void k_rowcol(const float* __restrict__ adj, float* rowsum, float* colsum) {
  __shared__ float sr[256], sc[256];
  int r = blockIdx.x, t = threadIdx.x;
  float a = 0.f, b = 0.f;
  for (int v = t; v < NN; v += 256) { a += adj[r * NN + v]; b += adj[v * NN + r]; }
  sr[t] = a; sc[t] = b; __syncthreads();
  for (int s = 128; s > 0; s >>= 1) {
    if (t < s) { sr[t] += sr[t + s]; sc[t] += sc[t + s]; }
    __syncthreads();
  }
  if (t == 0) { rowsum[r] = sr[0]; colsum[r] = sc[0]; }
}

// fwd = adj / rowsum, bwd = adjT / rowsum(adjT); stored row-major f16.
__global__ void k_norm_supp(const float* __restrict__ adj, const float* rowsum,
                            const float* colsum, _Float16* suppH) {
  int idx = blockIdx.x * blockDim.x + threadIdx.x;   // w*NN+v
  int w = idx >> 10, v = idx & 1023;
  suppH[idx]            = (_Float16)(adj[w * NN + v] / (rowsum[w] + 1e-8f));
  suppH[NN * NN + idx]  = (_Float16)(adj[v * NN + w] / (colsum[w] + 1e-8f));
}

// order-2 supports: C = A*A^T (symmetric), written into suppH[2+s].
__global__ void k_supp2(_Float16* suppH) {
  const _Float16* A = suppH + (size_t)blockIdx.y * NN * NN;
  _Float16*       C = suppH + (size_t)(2 + blockIdx.y) * NN * NN;
  int tm = blockIdx.x >> 4, tn = blockIdx.x & 15;
  int m0 = tm * 16, n0 = tn * 64;
  v8f acc[4];
  gemm_strip(A, NN, A, NN, m0, n0, NN, acc);   // A^T: Bc[n][k] = A[n][k]
  int lane = threadIdx.x & 31;
  int nl = lane & 15, mb = (lane & 16) ? 8 : 0;
#pragma unroll
  for (int j = 0; j < 4; ++j)
#pragma unroll
    for (int vr = 0; vr < 8; ++vr)
      C[(size_t)(m0 + mb + vr) * NN + n0 + 16 * j + nl] = (_Float16)acc[j][vr];
}

__global__ void k_prep_w(const float* Wr, const float* Wu, const float* Wc,
                         _Float16* Wpad) {
  int idx = blockIdx.x * blockDim.x + threadIdx.x;    // 3*64*352
  if (idx >= 3 * HID * KPAD) return;
  int g = idx / (HID * KPAD);
  int rem = idx - g * HID * KPAD;
  int o = rem / KPAD, k = rem % KPAD;
  const float* W = (g == 0) ? Wr : (g == 1) ? Wu : Wc;
  Wpad[idx] = (_Float16)((k < CCONV) ? W[o * CCONV + k] : 0.0f);
}

__global__ void k_init_h(const float* __restrict__ h0, float* h) {
  int idx = blockIdx.x * blockDim.x + threadIdx.x;    // 8*64*1024
  h[idx] = h0[idx % (HID * NN)];
}

__global__ void k_zero_pad(_Float16* zg, _Float16* zc) {
  int idx = blockIdx.x * blockDim.x + threadIdx.x;    // 8*1024*(KPAD-CCONV)
  if (idx >= BT * NN * (KPAD - CCONV)) return;
  int bn = idx / (KPAD - CCONV), k = idx % (KPAD - CCONV) + CCONV;
  zg[(size_t)bn * KPAD + k] = (_Float16)0.f;
  zc[(size_t)bn * KPAD + k] = (_Float16)0.f;
}

// --------------------------- per-step kernels ------------------------------
// Build z = concat([x_t, h]) in both GEMM-A layout [528][1024] and the
// node-major concatenated tensor [b][n][KPAD] (self channels 0..65).
__global__ void k_build_zg(const float* __restrict__ x, const float* __restrict__ h,
                           _Float16* zA, _Float16* zcatT, int t) {
  int idx = blockIdx.x * blockDim.x + threadIdx.x;    // 8*66*1024
  if (idx >= BT * CGATE * NN) return;
  int b = idx / (CGATE * NN);
  int rem = idx - b * CGATE * NN;
  int c = rem / NN, n = rem % NN;
  float v = (c < CIN)
      ? x[((size_t)(b * CIN + c) * NN + n) * STEPS + t]
      : h[(size_t)(b * HID + (c - CIN)) * NN + n];
  _Float16 hv = (_Float16)v;
  zA[(size_t)(b * CGATE + c) * NN + n] = hv;
  zcatT[((size_t)b * NN + n) * KPAD + c] = hv;
}

__global__ void k_build_zc(const float* __restrict__ x, const float* __restrict__ h,
                           const float* __restrict__ r,
                           _Float16* zA, _Float16* zcatT, int t) {
  int idx = blockIdx.x * blockDim.x + threadIdx.x;
  if (idx >= BT * CGATE * NN) return;
  int b = idx / (CGATE * NN);
  int rem = idx - b * CGATE * NN;
  int c = rem / NN, n = rem % NN;
  float v;
  if (c < CIN) {
    v = x[((size_t)(b * CIN + c) * NN + n) * STEPS + t];
  } else {
    size_t i = (size_t)(b * HID + (c - CIN)) * NN + n;
    v = r[i] * h[i];
  }
  _Float16 hv = (_Float16)v;
  zA[(size_t)(b * CGATE + c) * NN + n] = hv;
  zcatT[((size_t)b * NN + n) * KPAD + c] = hv;
}

// agg[bc,w] = sum_v z[bc,v] * supp[s][w][v].  A=[528][1024] row-major,
// B col-major view of supp^T is simply supp[w][v] row-major. Scatter results
// into zcatT channel (s+1)*66+c.  Grid: x = 33*16 strips, y = support.
__global__ void k_agg(const _Float16* __restrict__ zA,
                      const _Float16* __restrict__ suppH,
                      _Float16* zcatT) {
  int s = blockIdx.y;
  const _Float16* B = suppH + (size_t)s * NN * NN;
  int tm = blockIdx.x >> 4, tn = blockIdx.x & 15;
  int m0 = tm * 16, n0 = tn * 64;
  v8f acc[4];
  gemm_strip(zA, NN, B, NN, m0, n0, NN, acc);
  int lane = threadIdx.x & 31;
  int nl = lane & 15, mb = (lane & 16) ? 8 : 0;
#pragma unroll
  for (int j = 0; j < 4; ++j)
#pragma unroll
    for (int vr = 0; vr < 8; ++vr) {
      int m = m0 + mb + vr;                 // bc row
      int bb = m / CGATE, c = m - bb * CGATE;
      int w = n0 + 16 * j + nl;
      zcatT[((size_t)bb * NN + w) * KPAD + (s + 1) * CGATE + c] =
          (_Float16)acc[j][vr];
    }
}

// r and u gate convs in one launch: out[b,o,n] = sigmoid(Wg x Z + bg).
// Grid: x = 4*16 strips, y = batch, z = gate (0=r -> rbuf, 1=u -> ubuf).
__global__ void k_gate_ru(const _Float16* __restrict__ WpadR,
                          const _Float16* __restrict__ WpadU,
                          const _Float16* __restrict__ zcatT,
                          const float* __restrict__ br,
                          const float* __restrict__ bu,
                          float* rbuf, float* ubuf) {
  int b = blockIdx.y;
  int g = blockIdx.z;
  const _Float16* W    = g ? WpadU : WpadR;
  const float*    bias = g ? bu : br;
  float*          out  = g ? ubuf : rbuf;
  const _Float16* Bc = zcatT + (size_t)b * NN * KPAD;
  int tm = blockIdx.x >> 4, tn = blockIdx.x & 15;
  int m0 = tm * 16, n0 = tn * 64;
  v8f acc[4];
  gemm_strip(W, KPAD, Bc, KPAD, m0, n0, KPAD, acc);
  int lane = threadIdx.x & 31;
  int nl = lane & 15, mb = (lane & 16) ? 8 : 0;
#pragma unroll
  for (int j = 0; j < 4; ++j)
#pragma unroll
    for (int vr = 0; vr < 8; ++vr) {
      int o = m0 + mb + vr;
      int n = n0 + 16 * j + nl;
      float pre = acc[j][vr] + bias[o];
      out[((size_t)b * HID + o) * NN + n] = 1.0f / (1.0f + expf(-pre));
    }
}

// Candidate gate conv: cc[b,o,n] = tanh(Wc x Zc + bc).
__global__ void k_gate_c(const _Float16* __restrict__ W,
                         const _Float16* __restrict__ zcatT,
                         const float* __restrict__ bias,
                         float* out) {
  int b = blockIdx.y;
  const _Float16* Bc = zcatT + (size_t)b * NN * KPAD;
  int tm = blockIdx.x >> 4, tn = blockIdx.x & 15;
  int m0 = tm * 16, n0 = tn * 64;
  v8f acc[4];
  gemm_strip(W, KPAD, Bc, KPAD, m0, n0, KPAD, acc);
  int lane = threadIdx.x & 31;
  int nl = lane & 15, mb = (lane & 16) ? 8 : 0;
#pragma unroll
  for (int j = 0; j < 4; ++j)
#pragma unroll
    for (int vr = 0; vr < 8; ++vr) {
      int o = m0 + mb + vr;
      int n = n0 + 16 * j + nl;
      out[((size_t)b * HID + o) * NN + n] = tanhf(acc[j][vr] + bias[o]);
    }
}

// hn = u*h + (1-u)*c; GroupNorm(1 group) over (HID,NN) per batch; write new h,
// states output, and x_hat = Wo*hn + bo.  One block per batch, thread = node.
__global__ void k_update(float* h, const float* __restrict__ u,
                         const float* __restrict__ cc,
                         const float* __restrict__ gamma,
                         const float* __restrict__ beta,
                         const float* __restrict__ Wo,
                         const float* __restrict__ bo,
                         float* out, int t) {
  __shared__ float s1[1024], s2[1024];
  int b = blockIdx.x, n = threadIdx.x;
  float lsum = 0.f, lsq = 0.f;
  for (int c = 0; c < HID; ++c) {
    size_t i = ((size_t)b * HID + c) * NN + n;
    float uu = u[i];
    float hn = uu * h[i] + (1.0f - uu) * cc[i];
    lsum += hn; lsq += hn * hn;
  }
  s1[n] = lsum; s2[n] = lsq; __syncthreads();
  for (int st = 512; st > 0; st >>= 1) {
    if (n < st) { s1[n] += s1[n + st]; s2[n] += s2[n + st]; }
    __syncthreads();
  }
  const float inv = 1.0f / (float)(HID * NN);
  float mean = s1[0] * inv;
  float var  = s2[0] * inv - mean * mean;
  float rs = rsqrtf(var + 1e-5f);
  float a0 = 0.f, a1 = 0.f;
  for (int c = 0; c < HID; ++c) {
    size_t i = ((size_t)b * HID + c) * NN + n;
    float uu = u[i];
    float hn = uu * h[i] + (1.0f - uu) * cc[i];
    float g = (hn - mean) * rs * gamma[c] + beta[c];
    h[i] = g;
    out[IMP_SIZE + i * STEPS + t] = g;           // states[0][b][c][n][t]
    a0 += Wo[c] * g;
    a1 += Wo[HID + c] * g;
  }
  out[((size_t)(b * CIN + 0) * NN + n) * STEPS + t] = a0 + bo[0];
  out[((size_t)(b * CIN + 1) * NN + n) * STEPS + t] = a1 + bo[1];
}

// ------------------------------- launcher ----------------------------------
extern "C" void kernel_launch(void* const* d_in, const int* in_sizes, int n_in,
                              void* d_out, int out_size, void* d_ws, size_t ws_size,
                              hipStream_t stream) {
  const float* x     = (const float*)d_in[0];
  const float* adj   = (const float*)d_in[1];
  const float* h0    = (const float*)d_in[2];
  const float* Wr    = (const float*)d_in[3];
  const float* br    = (const float*)d_in[4];
  const float* Wu    = (const float*)d_in[5];
  const float* bu    = (const float*)d_in[6];
  const float* Wc    = (const float*)d_in[7];
  const float* bc_   = (const float*)d_in[8];
  const float* gamma = (const float*)d_in[9];
  const float* beta  = (const float*)d_in[10];
  const float* Wo    = (const float*)d_in[11];
  const float* bo    = (const float*)d_in[12];
  float* out = (float*)d_out;

  char* ws = (char*)d_ws;
  _Float16* suppH  = (_Float16*)(ws + OFF_SUPPH);
  float*    hstate = (float*)   (ws + OFF_HSTATE);
  _Float16* zgA    = (_Float16*)(ws + OFF_ZGA);
  _Float16* zcA    = (_Float16*)(ws + OFF_ZCA);
  _Float16* zcatG  = (_Float16*)(ws + OFF_ZCATG);
  _Float16* zcatC  = (_Float16*)(ws + OFF_ZCATC);
  float*    rbuf   = (float*)   (ws + OFF_R);
  float*    ubuf   = (float*)   (ws + OFF_U);
  float*    ccbuf  = (float*)   (ws + OFF_CC);
  _Float16* Wpad   = (_Float16*)(ws + OFF_WPAD);
  float*    rowsum = (float*)   (ws + OFF_ROWS);
  float*    colsum = (float*)   (ws + OFF_COLS);
  _Float16* WpadR = Wpad;
  _Float16* WpadU = Wpad + HID * KPAD;
  _Float16* WpadC = Wpad + 2 * HID * KPAD;

  // ---- one-time prep (supports, padded weights, h0 broadcast) ----
  k_rowcol<<<NN, 256, 0, stream>>>(adj, rowsum, colsum);
  k_norm_supp<<<(NN * NN) / 256, 256, 0, stream>>>(adj, rowsum, colsum, suppH);
  dim3 g2(1024, 2);
  k_supp2<<<g2, 32, 0, stream>>>(suppH);
  k_prep_w<<<(3 * HID * KPAD + 255) / 256, 256, 0, stream>>>(Wr, Wu, Wc, Wpad);
  k_init_h<<<(BT * HID * NN) / 256, 256, 0, stream>>>(h0, hstate);
  k_zero_pad<<<(BT * NN * (KPAD - CCONV) + 255) / 256, 256, 0, stream>>>(zcatG, zcatC);

  // ---- recurrent scan ----
  dim3 ga(33 * 16, NSUPP);      // aggregation strips
  dim3 gru(4 * 16, BT, 2);      // r+u gate GEMMs in one launch
  dim3 gc(4 * 16, BT);          // candidate gate GEMM
  const int nbz = (BT * CGATE * NN + 255) / 256;
  for (int t = 0; t < STEPS; ++t) {
    k_build_zg<<<nbz, 256, 0, stream>>>(x, hstate, zgA, zcatG, t);
    k_agg<<<ga, 32, 0, stream>>>(zgA, suppH, zcatG);
    k_gate_ru<<<gru, 32, 0, stream>>>(WpadR, WpadU, zcatG, br, bu, rbuf, ubuf);
    k_build_zc<<<nbz, 256, 0, stream>>>(x, hstate, rbuf, zcA, zcatC, t);
    k_agg<<<ga, 32, 0, stream>>>(zcA, suppH, zcatC);
    k_gate_c<<<gc, 32, 0, stream>>>(WpadC, zcatC, bc_, ccbuf);
    k_update<<<BT, 1024, 0, stream>>>(hstate, ubuf, ccbuf, gamma, beta, Wo, bo, out, t);
  }
}
